// MaxMarginContrastive_Loss_50113678410453
// MI455X (gfx1250) — matrix operand order repaired and checked by reading, not compiled, not run on hardware
//
#include <hip/hip_runtime.h>
#include <hip/hip_bf16.h>
#include <math.h>

#define EMBED    1024
#define N_IMG    32
#define N_REG    36
#define N_CAP    32
#define MAX_W    32
#define EPSV     1e-8f
#define LAMBDA_SM 9.0f
#define MARGIN   0.2f

typedef __bf16 bf16;
typedef __attribute__((ext_vector_type(16))) __bf16 v16bf;
typedef __attribute__((ext_vector_type(4)))  __bf16 bf16x4;
typedef __attribute__((ext_vector_type(8)))  float  v8f;

static __device__ __forceinline__ float warp_reduce_sum(float v) {
    #pragma unroll
    for (int off = 16; off > 0; off >>= 1)
        v += __shfl_down(v, off, 32);
    return v;
}

// ---------------------------------------------------------------------------
// Init: zero score, convert weights to bf16, broadcast cap_emb -> q0 (bf16)
// ---------------------------------------------------------------------------
__global__ void k_init(float* __restrict__ score,
                       bf16* __restrict__ wlb, bf16* __restrict__ wgb,
                       const float* __restrict__ w_lin, const float* __restrict__ w_gate,
                       bf16* __restrict__ qbuf, const float* __restrict__ cap_emb) {
    long idx    = (long)blockIdx.x * blockDim.x + threadIdx.x;
    long stride = (long)gridDim.x * blockDim.x;
    for (long t = idx; t < N_CAP * N_IMG; t += stride) score[t] = 0.f;
    const long WN = (long)EMBED * 2 * EMBED;
    for (long t = idx; t < WN; t += stride) {
        wlb[t] = (bf16)w_lin[t];
        wgb[t] = (bf16)w_gate[t];
    }
    const long QN = (long)N_CAP * N_IMG * MAX_W * EMBED;
    for (long t = idx; t < QN; t += stride) {
        long d = t & (EMBED - 1);
        long m = t >> 10;          // ((c*32)+i)*32 + w
        long w = m & 31;
        long c = m >> 10;
        qbuf[t] = (bf16)cap_emb[(c * MAX_W + w) * EMBED + d];
    }
}

// ---------------------------------------------------------------------------
// Caption word norms: w1[c*32+w] = ||cap_emb[c,w,:]||
// ---------------------------------------------------------------------------
__global__ void k_capnorm(const float* __restrict__ cap_emb, float* __restrict__ w1) {
    int row  = blockIdx.x * 8 + (threadIdx.x >> 5);   // 128 blocks * 8 waves = 1024 rows
    int lane = threadIdx.x & 31;
    if (row >= N_CAP * MAX_W) return;
    const float* p = cap_emb + (long)row * EMBED;
    float s = 0.f;
    for (int k = lane; k < EMBED; k += 32) { float v = p[k]; s += v * v; }
    s = warp_reduce_sum(s);
    if (lane == 0) w1[row] = sqrtf(s);
}

// ---------------------------------------------------------------------------
// Attention + weighted context + cosine/score. One block per (c,i) pair.
// ---------------------------------------------------------------------------
__global__ void __launch_bounds__(256) k_attn(
        const float* __restrict__ img_emb, const bf16* __restrict__ qbuf,
        const float* __restrict__ cap_emb, const int* __restrict__ cap_lens,
        const float* __restrict__ w1, bf16* __restrict__ wcbuf,
        float* __restrict__ score) {
    __shared__ float attn_s[N_REG][MAX_W + 1];
    __shared__ float s_score;

    const int p = blockIdx.x;
    const int c = p >> 5, i = p & 31;
    const int tid = threadIdx.x, lane = tid & 31, wv = tid >> 5;
    const int len = cap_lens[c];
    const float* img  = img_emb + (long)i * N_REG * EMBED;
    const bf16*  qrow = qbuf + (long)p * MAX_W * EMBED;

    if (tid == 0) s_score = 0.f;

    // ---- phase 1: attn[r][w] = img[r,:] . q[w,:]  (wave per dot, lane over d)
    for (int j = wv; j < N_REG * MAX_W; j += 8) {
        const int r = j >> 5, w = j & 31;
        const float* ip = img + (long)r * EMBED;
        const bf16*  qp = qrow + (long)w * EMBED;
        float acc = 0.f;
        #pragma unroll
        for (int k = 0; k < 8; ++k) {
            int d = (k * 32 + lane) * 4;
            float4 iv = *(const float4*)(ip + d);
            bf16x4 qv = *(const bf16x4*)(qp + d);
            acc += iv.x * (float)qv.x + iv.y * (float)qv.y +
                   iv.z * (float)qv.z + iv.w * (float)qv.w;
        }
        acc = warp_reduce_sum(acc);
        if (lane == 0) attn_s[r][w] = acc;
    }
    __syncthreads();

    // ---- phase 2a: leaky relu + masked L2 norm over w (per row r)
    if (tid < N_REG) {
        const int r = tid;
        float ss = 0.f;
        #pragma unroll
        for (int w = 0; w < MAX_W; ++w) {
            float v = attn_s[r][w];
            v = v > 0.f ? v : 0.1f * v;
            attn_s[r][w] = v;
            if (w < len) ss += v * v;
        }
        float inv = 1.f / (sqrtf(ss) + EPSV);
        #pragma unroll
        for (int w = 0; w < MAX_W; ++w) attn_s[r][w] *= inv;
    }
    __syncthreads();

    // ---- phase 2b: softmax over r (per column w), temperature LAMBDA_SM
    if (tid < MAX_W) {
        const int w = tid;
        float vmax = -1e30f;
        #pragma unroll
        for (int r = 0; r < N_REG; ++r) vmax = fmaxf(vmax, attn_s[r][w]);
        float ssum = 0.f;
        #pragma unroll
        for (int r = 0; r < N_REG; ++r) {
            float e = __expf(LAMBDA_SM * (attn_s[r][w] - vmax));
            attn_s[r][w] = e;
            ssum += e;
        }
        float inv = 1.f / ssum;
        #pragma unroll
        for (int r = 0; r < N_REG; ++r) attn_s[r][w] *= inv;
    }
    __syncthreads();

    // ---- phase 3: wc[w,d] = sum_r sm[r][w]*img[r,d]; cosine + masked score
    const int wbase = wv * 4;   // each wave owns 4 caption words
    float p12[4] = {0.f, 0.f, 0.f, 0.f};
    float p22[4] = {0.f, 0.f, 0.f, 0.f};
    for (int k = 0; k < 32; ++k) {
        const int d = k * 32 + lane;
        float acc[4] = {0.f, 0.f, 0.f, 0.f};
        for (int r = 0; r < N_REG; ++r) {
            float iv = img[(long)r * EMBED + d];
            #pragma unroll
            for (int j = 0; j < 4; ++j) acc[j] += attn_s[r][wbase + j] * iv;
        }
        #pragma unroll
        for (int j = 0; j < 4; ++j) {
            const int w = wbase + j;
            wcbuf[((long)p * MAX_W + w) * EMBED + d] = (bf16)acc[j];
            float cv = cap_emb[((long)c * MAX_W + w) * EMBED + d];
            p12[j] += cv * acc[j];
            p22[j] += acc[j] * acc[j];
        }
    }
    #pragma unroll
    for (int j = 0; j < 4; ++j) {
        p12[j] = warp_reduce_sum(p12[j]);
        p22[j] = warp_reduce_sum(p22[j]);
    }
    if (lane == 0) {
        float local = 0.f;
        #pragma unroll
        for (int j = 0; j < 4; ++j) {
            const int w = wbase + j;
            float wnorm = sqrtf(p22[j]);
            float cosv  = p12[j] / fmaxf(w1[c * MAX_W + w] * wnorm, EPSV);
            if (w < len) local += cosv;
        }
        atomicAdd(&s_score, local);
    }
    __syncthreads();
    if (tid == 0) score[p] += s_score / (float)len;
}

// ---------------------------------------------------------------------------
// Gating GEMM: h = tanh(cat@Wl^T+bl), g = sigmoid(cat@Wg^T+bg),
//              qnew = q*g + h*(1-g).  cat = [q | wc], M=32768 K=2048 N=1024.
// Block tile 128x64, 8 waves each 32x32, WMMA bf16 16x16x32,
// double-buffered LDS, uniform per-step q/wc base select.
// ---------------------------------------------------------------------------
#define TM 128
#define TN 64
#define TK 32
#define NSTEP 64            // 2048 / TK
#define LDA 40              // padded LDS row (bf16 elements)

__global__ void __launch_bounds__(256) k_gate(
        const bf16* __restrict__ qbuf, const bf16* __restrict__ wcbuf,
        const bf16* __restrict__ wl,   const bf16* __restrict__ wg,
        const float* __restrict__ b_lin, const float* __restrict__ b_gate,
        bf16* __restrict__ qnew) {
    __shared__ bf16 As [2][TM][LDA];
    __shared__ bf16 Bls[2][TN][LDA];
    __shared__ bf16 Bgs[2][TN][LDA];

    const int nb = blockIdx.x;             // 0..15
    const int mb = blockIdx.y;             // 0..255
    const int tid = threadIdx.x, lane = tid & 31, wv = tid >> 5;
    const int wr = wv >> 1, wn = wv & 1;   // 4x2 wave grid
    const int lr = lane & 15;
    const int kbA = (lane >> 4) * 16;
    const int kbB = (lane >> 4) * 16;
    const long m0 = (long)mb * TM;
    const int  n0 = nb * TN;

    // fixed staging assignment: thread stages A rows (arow, arow+64) and
    // B row brow, all at bf16 column offset ko (8 elements = 16 bytes).
    const int arow = tid >> 2;
    const int brow = tid >> 2;
    const int ko   = (tid & 3) * 8;

    v8f hAcc[2][2] = {};
    v8f gAcc[2][2] = {};

    uint4 ra0, ra1, rbl, rbg;

    // prologue: stage step 0 into buffer 0 (step 0 reads the q half)
    {
        const bf16* Ab = qbuf;
        ra0 = *(const uint4*)(Ab + (m0 + arow) * EMBED + ko);
        ra1 = *(const uint4*)(Ab + (m0 + arow + 64) * EMBED + ko);
        const long wko = (long)(n0 + brow) * (2 * EMBED) + ko;
        rbl = *(const uint4*)(wl + wko);
        rbg = *(const uint4*)(wg + wko);
        *(uint4*)&As [0][arow][ko]      = ra0;
        *(uint4*)&As [0][arow + 64][ko] = ra1;
        *(uint4*)&Bls[0][brow][ko]      = rbl;
        *(uint4*)&Bgs[0][brow][ko]      = rbg;
    }

    for (int s = 0; s < NSTEP; ++s) {
        __syncthreads();
        const int  buf  = s & 1;
        const bool more = (s + 1) < NSTEP;

        // issue next step's global loads early (hide under WMMA)
        if (more) {
            const int t = s + 1;
            const int klocal = (t & 31) * TK;                       // uniform
            const bf16* Ab = (t < NSTEP / 2) ? qbuf : wcbuf;        // uniform scalar select
            ra0 = *(const uint4*)(Ab + (m0 + arow) * EMBED + klocal + ko);
            ra1 = *(const uint4*)(Ab + (m0 + arow + 64) * EMBED + klocal + ko);
            const long wko = (long)(n0 + brow) * (2 * EMBED) + (long)t * TK + ko;
            rbl = *(const uint4*)(wl + wko);
            rbg = *(const uint4*)(wg + wko);
        }
        // prefetch weight stream a few steps ahead into near caches
        if (s + 3 < NSTEP) {
            const long pko = (long)(n0 + brow) * (2 * EMBED) + (long)(s + 3) * TK + ko;
            __builtin_prefetch(wl + pko, 0, 0);
            __builtin_prefetch(wg + pko, 0, 0);
        }

        // load fragments from current buffer
        v16bf a[2], bl[2], bg[2];
        #pragma unroll
        for (int rg = 0; rg < 2; ++rg) {
            const bf16* pa = &As[buf][wr * 32 + rg * 16 + lr][kbA];
            #pragma unroll
            for (int t = 0; t < 8; ++t) { a[rg][t] = pa[t]; a[rg][8 + t] = pa[8 + t]; }
        }
        #pragma unroll
        for (int cg = 0; cg < 2; ++cg) {
            const bf16* pl = &Bls[buf][wn * 32 + cg * 16 + lr][kbB];
            const bf16* pg = &Bgs[buf][wn * 32 + cg * 16 + lr][kbB];
            #pragma unroll
            for (int t = 0; t < 16; ++t) { bl[cg][t] = pl[t]; bg[cg][t] = pg[t]; }
        }

        #pragma unroll
        for (int rg = 0; rg < 2; ++rg)
            #pragma unroll
            for (int cg = 0; cg < 2; ++cg) {
                hAcc[rg][cg] = __builtin_amdgcn_wmma_f32_16x16x32_bf16(
                    false, a[rg], false, bl[cg], (short)0, hAcc[rg][cg], false, false);
                gAcc[rg][cg] = __builtin_amdgcn_wmma_f32_16x16x32_bf16(
                    false, a[rg], false, bg[cg], (short)0, gAcc[rg][cg], false, false);
            }

        // store next step's tiles into the other buffer
        if (more) {
            const int nxt = (s + 1) & 1;
            *(uint4*)&As [nxt][arow][ko]      = ra0;
            *(uint4*)&As [nxt][arow + 64][ko] = ra1;
            *(uint4*)&Bls[nxt][brow][ko]      = rbl;
            *(uint4*)&Bgs[nxt][brow][ko]      = rbg;
        }
    }

    // epilogue: bias + tanh/sigmoid + gated update
    const int mlo = (lane >> 4) * 8;
    #pragma unroll
    for (int rg = 0; rg < 2; ++rg)
        #pragma unroll
        for (int cg = 0; cg < 2; ++cg) {
            const int n = n0 + wn * 32 + cg * 16 + lr;
            const float bl_ = b_lin[n];
            const float bg_ = b_gate[n];
            #pragma unroll
            for (int v = 0; v < 8; ++v) {
                const long m = m0 + wr * 32 + rg * 16 + mlo + v;
                float hh = tanhf(hAcc[rg][cg][v] + bl_);
                float gg = 1.f / (1.f + __expf(-(gAcc[rg][cg][v] + bg_)));
                float qo = (float)qbuf[m * EMBED + n];
                qnew[m * EMBED + n] = (bf16)(qo * gg + hh * (1.f - gg));
            }
        }
}

// ---------------------------------------------------------------------------
// Final max-margin loss from 32x32 score matrix
// ---------------------------------------------------------------------------
__global__ void k_loss(const float* __restrict__ score, float* __restrict__ out) {
    __shared__ float s[N_CAP][N_IMG + 1];
    __shared__ float acc[64];
    const int tid = threadIdx.x;
    for (int t = tid; t < N_CAP * N_IMG; t += 256) s[t >> 5][t & 31] = score[t];
    __syncthreads();
    float v = 0.f;
    if (tid < 32) {                       // cost_s: row max
        const int c = tid;
        const float diag = s[c][c];
        float mx = 0.f;
        for (int i = 0; i < N_IMG; ++i)
            if (i != c) mx = fmaxf(mx, fmaxf(0.f, MARGIN + s[c][i] - diag));
        v = mx;
    } else if (tid < 64) {                // cost_im: column max
        const int i = tid - 32;
        const float diag = s[i][i];
        float mx = 0.f;
        for (int c = 0; c < N_CAP; ++c)
            if (c != i) mx = fmaxf(mx, fmaxf(0.f, MARGIN + s[c][i] - diag));
        v = mx;
    }
    if (tid < 64) acc[tid] = v;
    __syncthreads();
    if (tid == 0) {
        float t = 0.f;
        for (int j = 0; j < 64; ++j) t += acc[j];
        out[0] = t;
    }
}

// ---------------------------------------------------------------------------
extern "C" void kernel_launch(void* const* d_in, const int* in_sizes, int n_in,
                              void* d_out, int out_size, void* d_ws, size_t ws_size,
                              hipStream_t stream) {
    (void)in_sizes; (void)n_in; (void)out_size; (void)ws_size;
    const float* img_emb  = (const float*)d_in[0];
    const float* cap_emb  = (const float*)d_in[1];
    const int*   cap_lens = (const int*)d_in[2];
    const float* w_lin    = (const float*)d_in[3];
    const float* b_lin    = (const float*)d_in[4];
    const float* w_gate   = (const float*)d_in[5];
    const float* b_gate   = (const float*)d_in[6];
    float* out = (float*)d_out;

    char* ws = (char*)d_ws;
    size_t off = 0;
    auto alloc = [&](size_t bytes) -> void* {
        void* p = ws + off;
        off += (bytes + 255) & ~(size_t)255;
        return p;
    };
    const size_t QB = (size_t)N_CAP * N_IMG * MAX_W * EMBED * sizeof(bf16);
    bf16* qA   = (bf16*)alloc(QB);
    bf16* qB   = (bf16*)alloc(QB);
    bf16* wc   = (bf16*)alloc(QB);
    bf16* wlb  = (bf16*)alloc((size_t)EMBED * 2 * EMBED * sizeof(bf16));
    bf16* wgb  = (bf16*)alloc((size_t)EMBED * 2 * EMBED * sizeof(bf16));
    float* w1    = (float*)alloc(N_CAP * MAX_W * sizeof(float));
    float* score = (float*)alloc(N_CAP * N_IMG * sizeof(float));

    hipLaunchKernelGGL(k_init, dim3(2048), dim3(256), 0, stream,
                       score, wlb, wgb, w_lin, w_gate, qA, cap_emb);
    hipLaunchKernelGGL(k_capnorm, dim3(128), dim3(256), 0, stream, cap_emb, w1);

    bf16* qcur = qA;
    bf16* qnext = qB;
    for (int it = 0; it < 3; ++it) {
        hipLaunchKernelGGL(k_attn, dim3(N_CAP * N_IMG), dim3(256), 0, stream,
                           img_emb, qcur, cap_emb, cap_lens, w1, wc, score);
        if (it < 2) {   // last q-update is dead code w.r.t. the loss
            hipLaunchKernelGGL(k_gate, dim3(16, 256), dim3(256), 0, stream,
                               qcur, wc, wlb, wgb, b_lin, b_gate, qnext);
            bf16* t = qcur; qcur = qnext; qnext = t;
        }
    }
    hipLaunchKernelGGL(k_loss, dim3(1), dim3(256), 0, stream, score, out);
}
